// Mamba1_19679540150392
// MI455X (gfx1250) — matrix operand order repaired and checked by reading, not compiled, not run on hardware
//
#include <hip/hip_runtime.h>

#define SEQ    2048
#define DMODEL 1024
#define DINNER 2048
#define NXZ    4096
#define NSTATE 16
#define RANKDT 64
#define PROJN  96
#define TCHUNK 128

// LDS tile geometry for the WMMA GEMM: block tile 128(M) x 128(N), K-step 32.
// Row stride 40 halfs (80B = 20 banks) -> conflict-free 16-lane row-strided reads.
#define KSTEP   32
#define LDSROW  40
#define BLK     128

typedef __bf16 bf16_t;
typedef __attribute__((ext_vector_type(16))) __bf16 v16bf;
typedef __attribute__((ext_vector_type(8)))  __bf16 v8bf;
typedef __attribute__((ext_vector_type(8)))  float  v8f;

#if defined(__gfx1250__) && __has_builtin(__builtin_amdgcn_global_load_async_to_lds_b128) && __has_builtin(__builtin_amdgcn_s_wait_asynccnt)
#define USE_ASYNC 1
// Parameter types per hipcc diagnostic: (v4i global*, v4i shared*, Ii, Ii)
typedef int v4i_vs __attribute__((vector_size(4 * sizeof(int))));
typedef v4i_vs __attribute__((address_space(1)))* gbl_v4i_ptr;
typedef v4i_vs __attribute__((address_space(3)))* lds_v4i_ptr;
#else
#define USE_ASYNC 0
#endif

static __device__ __forceinline__ bf16_t f2bf(float f) {
  unsigned u = __builtin_bit_cast(unsigned, f);
  u += 0x7FFFu + ((u >> 16) & 1u);          // round-to-nearest-even
  unsigned short s = (unsigned short)(u >> 16);
  return __builtin_bit_cast(bf16_t, s);
}

// ---------------------------------------------------------------- converts
__global__ void mb1_cvt_bf16(const float* __restrict__ src,
                             bf16_t* __restrict__ dst, int n) {
  int i = blockIdx.x * blockDim.x + threadIdx.x;
  int stride = gridDim.x * blockDim.x;
  for (; i < n; i += stride) dst[i] = f2bf(src[i]);
}

// --------------------------------------------- WMMA fragments from LDS tile
// A (16x32 MxK), 16-bit layout: lane<16 -> row=lane, K in {0..7, 16..23};
// lane>=16 -> row=lane-16, K in {8..15, 24..31}.
static __device__ __forceinline__ v16bf lds_fragA(const bf16_t* base) {
  int lane = threadIdx.x & 31;
  const bf16_t* p = base + (lane & 15) * LDSROW + ((lane & 16) ? 8 : 0);
  v8bf lo = *(const v8bf*)p;          // K = koff .. koff+7
  v8bf hi = *(const v8bf*)(p + 16);   // K = koff+16 .. koff+23
  v16bf f;
#pragma unroll
  for (int i = 0; i < 8; ++i) { f[i] = lo[i]; f[i + 8] = hi[i]; }
  return f;
}

// B (32x16 KxN) with B[k][n] = W[n][k]:
// lane<16 -> col n=lane, K in {0..7, 16..23}; lane>=16 -> K in {8..15, 24..31}.
static __device__ __forceinline__ v16bf lds_fragB(const bf16_t* base) {
  int lane = threadIdx.x & 31;
  const bf16_t* p = base + (lane & 15) * LDSROW + ((lane & 16) ? 8 : 0);
  v8bf lo = *(const v8bf*)p;
  v8bf hi = *(const v8bf*)(p + 16);
  v16bf f;
#pragma unroll
  for (int i = 0; i < 8; ++i) { f[i] = lo[i]; f[i + 8] = hi[i]; }
  return f;
}

// C = A(MxK) @ W(NxK)^T, row-major, fp32 out.
// Block: 256 threads = 8 waves (4 in M x 2 in N); wave tile 32x64 -> 8 WMMA/K-step.
// Global->LDS staging is double-buffered via async-to-LDS (ASYNCcnt) when available.
__global__ __launch_bounds__(256) void mb1_gemm_bf16_wmma(
    const bf16_t* __restrict__ A, const bf16_t* __restrict__ W,
    float* __restrict__ C, int M, int N, int K) {
  __shared__ __align__(16) bf16_t sA[2][BLK * LDSROW];
  __shared__ __align__(16) bf16_t sB[2][BLK * LDSROW];
  const int tid  = threadIdx.x;
  const int wave = tid >> 5;
  const int lane = tid & 31;
  const int m0b = blockIdx.y * BLK;
  const int n0b = blockIdx.x * BLK;
  const int wm = (wave & 3) * 32;    // wave M offset in block tile
  const int wn = (wave >> 2) * 64;   // wave N offset in block tile

  v8f acc[2][4];
#pragma unroll
  for (int mi = 0; mi < 2; ++mi)
#pragma unroll
    for (int ni = 0; ni < 4; ++ni) acc[mi][ni] = v8f{};

  // Stage one 128x32 A tile + 128x32 B tile: 1024 x 16B chunks, 4 per thread.
  auto stage = [&](int buf, int k0) {
#pragma unroll
    for (int c = 0; c < 2; ++c) {
      int idx = tid * 2 + c;                 // 0..511
      int r = idx >> 2, s = idx & 3;         // row, 16B segment
      const bf16_t* ga = A + (size_t)(m0b + r) * K + k0 + s * 8;
      const bf16_t* gb = W + (size_t)(n0b + r) * K + k0 + s * 8;
      bf16_t* la = &sA[buf][r * LDSROW + s * 8];
      bf16_t* lb = &sB[buf][r * LDSROW + s * 8];
#if USE_ASYNC
      __builtin_amdgcn_global_load_async_to_lds_b128(
          (gbl_v4i_ptr)(unsigned long long)ga,
          (lds_v4i_ptr)(unsigned)(unsigned long long)la, 0, 0);
      __builtin_amdgcn_global_load_async_to_lds_b128(
          (gbl_v4i_ptr)(unsigned long long)gb,
          (lds_v4i_ptr)(unsigned)(unsigned long long)lb, 0, 0);
#else
      *(v8bf*)la = *(const v8bf*)ga;
      *(v8bf*)lb = *(const v8bf*)gb;
#endif
    }
  };

  const int nk = K / KSTEP;
  stage(0, 0);
#if USE_ASYNC
  __builtin_amdgcn_s_wait_asynccnt(0);
#endif
  __syncthreads();

  int buf = 0;
  for (int kt = 0; kt < nk; ++kt) {
    if (kt + 1 < nk) stage(buf ^ 1, (kt + 1) * KSTEP);   // overlap with compute

    v16bf af[2], bfv[4];
#pragma unroll
    for (int mi = 0; mi < 2; ++mi)
      af[mi] = lds_fragA(&sA[buf][(wm + mi * 16) * LDSROW]);
#pragma unroll
    for (int ni = 0; ni < 4; ++ni)
      bfv[ni] = lds_fragB(&sB[buf][(wn + ni * 16) * LDSROW]);
#pragma unroll
    for (int mi = 0; mi < 2; ++mi)
#pragma unroll
      for (int ni = 0; ni < 4; ++ni)
        acc[mi][ni] = __builtin_amdgcn_wmma_f32_16x16x32_bf16(
            false, af[mi], false, bfv[ni], (short)0, acc[mi][ni], false, false);

    if (kt + 1 < nk) {
#if USE_ASYNC
      __builtin_amdgcn_s_wait_asynccnt(0);
#endif
      __syncthreads();
      buf ^= 1;
    }
  }

  // C/D layout: VGPR v -> row +v (lanes<16) / +v+8 (lanes>=16), col = lane&15
  int col = lane & 15;
  int rhi = (lane & 16) ? 8 : 0;
#pragma unroll
  for (int mi = 0; mi < 2; ++mi) {
    float* Cb = C + (size_t)(m0b + wm + mi * 16 + rhi) * N + n0b + wn + col;
#pragma unroll
    for (int v = 0; v < 8; ++v)
#pragma unroll
      for (int ni = 0; ni < 4; ++ni)
        Cb[(size_t)v * N + ni * 16] = acc[mi][ni][v];
  }
}

// ------------------------------------------------- depthwise conv4 + SiLU
__global__ void mb1_conv_silu(const float* __restrict__ xz,
                              const float* __restrict__ cw,
                              const float* __restrict__ cb,
                              float* __restrict__ u) {
  int idx = blockIdx.x * blockDim.x + threadIdx.x;   // t*DINNER + d
  if (idx >= SEQ * DINNER) return;
  int t = idx / DINNER, d = idx - t * DINNER;
  float acc = cb[d];
#pragma unroll
  for (int j = 0; j < 4; ++j) {
    int ts = t + j - 3;
    if (ts >= 0) acc += xz[(size_t)ts * NXZ + d] * cw[d * 4 + j];
  }
  u[idx] = acc / (1.0f + __expf(-acc));              // SiLU
}

// ------------------------------------- projt = u @ W_x^T (N=96, exact fp32)
__global__ __launch_bounds__(128) void mb1_proj_x(const float* __restrict__ u,
                                                  const float* __restrict__ Wx,
                                                  float* __restrict__ projt) {
  __shared__ float su[DINNER];
  int t = blockIdx.x;
  for (int i = threadIdx.x; i < DINNER; i += 128)
    su[i] = u[(size_t)t * DINNER + i];
  __syncthreads();
  int c = threadIdx.x;
  if (c < PROJN) {
    const float* w = Wx + (size_t)c * DINNER;
    float acc = 0.f;
#pragma unroll 8
    for (int k = 0; k < DINNER; ++k) acc += su[k] * w[k];
    projt[t * PROJN + c] = acc;
  }
}

// ----------------------- delta = softplus(dt_lr @ W_dt^T + dt_bias) (K=64)
__global__ __launch_bounds__(256) void mb1_delta(const float* __restrict__ projt,
                                                 const float* __restrict__ Wdt,
                                                 const float* __restrict__ dtb,
                                                 float* __restrict__ delta) {
  __shared__ float sp[RANKDT];
  int t = blockIdx.x;
  if (threadIdx.x < RANKDT) sp[threadIdx.x] = projt[t * PROJN + threadIdx.x];
  __syncthreads();
  for (int d = threadIdx.x; d < DINNER; d += 256) {
    const float* w = Wdt + (size_t)d * RANKDT;
    float acc = dtb[d];
#pragma unroll 8
    for (int r = 0; r < RANKDT; ++r) acc += sp[r] * w[r];
    delta[(size_t)t * DINNER + d] = (acc > 20.f) ? acc : log1pf(__expf(acc));
  }
}

// --------- fused selective scan + D*u skip + SiLU(gate) + bf16 down-convert
__global__ __launch_bounds__(256) void mb1_scan(
    const float* __restrict__ delta, const float* __restrict__ u,
    const float* __restrict__ projt, const float* __restrict__ A_log,
    const float* __restrict__ Dskip, const float* __restrict__ xz,
    bf16_t* __restrict__ ybf) {
  __shared__ float Bs[TCHUNK * NSTATE];
  __shared__ float Cs[TCHUNK * NSTATE];
  int d = blockIdx.x * 256 + threadIdx.x;            // channel, 2048 total
  float Aneg[NSTATE], h[NSTATE];
#pragma unroll
  for (int n = 0; n < NSTATE; ++n) {
    Aneg[n] = -__expf(A_log[d * NSTATE + n]);        // A = -exp(A_log)
    h[n] = 0.f;
  }
  float Dsk = Dskip[d];
  for (int tc = 0; tc < SEQ; tc += TCHUNK) {
    __syncthreads();
    for (int i = threadIdx.x; i < TCHUNK * NSTATE; i += 256) {
      int tt = i >> 4, n = i & 15;
      Bs[i] = projt[(size_t)(tc + tt) * PROJN + RANKDT + n];
      Cs[i] = projt[(size_t)(tc + tt) * PROJN + RANKDT + NSTATE + n];
    }
    __syncthreads();
    for (int tt = 0; tt < TCHUNK; ++tt) {
      int t = tc + tt;
      float dl = delta[(size_t)t * DINNER + d];
      float uv = u[(size_t)t * DINNER + d];
      float du = dl * uv;
      float y = 0.f;
#pragma unroll
      for (int n = 0; n < NSTATE; ++n) {
        float dA = __expf(dl * Aneg[n]);
        h[n] = dA * h[n] + du * Bs[tt * NSTATE + n];
        y += h[n] * Cs[tt * NSTATE + n];
      }
      y += Dsk * uv;
      float g = xz[(size_t)t * NXZ + DINNER + d];    // gate half of xz
      y *= g / (1.f + __expf(-g));                   // * SiLU(gate)
      ybf[(size_t)t * DINNER + d] = f2bf(y);
    }
  }
}

// --------------------------------------------------------------- dispatch
extern "C" void kernel_launch(void* const* d_in, const int* in_sizes, int n_in,
                              void* d_out, int out_size, void* d_ws, size_t ws_size,
                              hipStream_t stream) {
  (void)in_sizes; (void)n_in; (void)out_size; (void)ws_size;
  const float* H    = (const float*)d_in[0];   // (1, 2048, 1024)
  const float* Win  = (const float*)d_in[1];   // (4096, 1024)
  const float* cw   = (const float*)d_in[2];   // (2048, 4)
  const float* cb   = (const float*)d_in[3];   // (2048,)
  const float* Wx   = (const float*)d_in[4];   // (96, 2048)
  const float* Wdt  = (const float*)d_in[5];   // (2048, 64)
  const float* dtb  = (const float*)d_in[6];   // (2048,)
  const float* Alog = (const float*)d_in[7];   // (2048, 16)
  const float* Dsk  = (const float*)d_in[8];   // (2048,)
  const float* Wout = (const float*)d_in[9];   // (1024, 2048)
  float* out = (float*)d_out;                  // (1, 2048, 1024) fp32

  char* ws = (char*)d_ws;
  size_t off = 0;
  auto take = [&](size_t bytes) -> char* {
    char* p = ws + off;
    off += (bytes + 255) & ~(size_t)255;
    return p;
  };
  float*  xz     = (float*) take((size_t)SEQ * NXZ * 4);       // 32 MB
  bf16_t* Hbf    = (bf16_t*)take((size_t)SEQ * DMODEL * 2);    //  4 MB
  bf16_t* Winbf  = (bf16_t*)take((size_t)NXZ * DMODEL * 2);    //  8 MB
  bf16_t* Woutbf = (bf16_t*)take((size_t)DMODEL * DINNER * 2); //  4 MB
  float*  u      = (float*) take((size_t)SEQ * DINNER * 4);    // 16 MB
  float*  projt  = (float*) take((size_t)SEQ * PROJN * 4);     // .75 MB
  float*  delta  = (float*) take((size_t)SEQ * DINNER * 4);    // 16 MB
  bf16_t* ybf    = (bf16_t*)take((size_t)SEQ * DINNER * 2);    //  8 MB

  // 1) bf16 copies for WMMA GEMMs
  mb1_cvt_bf16<<<1024, 256, 0, stream>>>(H,    Hbf,    SEQ * DMODEL);
  mb1_cvt_bf16<<<1024, 256, 0, stream>>>(Win,  Winbf,  NXZ * DMODEL);
  mb1_cvt_bf16<<<1024, 256, 0, stream>>>(Wout, Woutbf, DMODEL * DINNER);

  // 2) xz = H @ W_in^T   (M=2048, N=4096, K=1024)
  mb1_gemm_bf16_wmma<<<dim3(NXZ / BLK, SEQ / BLK), 256, 0, stream>>>(
      Hbf, Winbf, xz, SEQ, NXZ, DMODEL);

  // 3) depthwise causal conv + SiLU -> u
  mb1_conv_silu<<<(SEQ * DINNER) / 256, 256, 0, stream>>>(xz, cw, cb, u);

  // 4) projt = u @ W_x^T  (N=96)
  mb1_proj_x<<<SEQ, 128, 0, stream>>>(u, Wx, projt);

  // 5) delta = softplus(dt_lr @ W_dt^T + dt_bias)
  mb1_delta<<<SEQ, 256, 0, stream>>>(projt, Wdt, dtb, delta);

  // 6) selective scan + skip + gate, bf16 output
  mb1_scan<<<DINNER / 256, 256, 0, stream>>>(delta, u, projt, Alog, Dsk, xz, ybf);

  // 7) out = y @ W_out^T  (M=2048, N=1024, K=2048)
  mb1_gemm_bf16_wmma<<<dim3(DMODEL / BLK, SEQ / BLK), 256, 0, stream>>>(
      ybf, Woutbf, out, SEQ, DMODEL, DINNER);
}
